// Decoder_13168369729862
// MI455X (gfx1250) — matrix operand order repaired and checked by reading, not compiled, not run on hardware
//
#include <hip/hip_runtime.h>
#include <hip/hip_bf16.h>

// Problem dims (from reference)
#define kB 32
#define kS 128
#define kE 512
#define kH 512
#define kD 256
#define kA 256
#define kV 32000
#define kT 48
#define kCat (kD + kE + kH)   // 1280, RNN input = [x_t | context | h_prev]

typedef __attribute__((ext_vector_type(16))) __bf16 v16bf;
typedef __attribute__((ext_vector_type(8)))  __bf16 v8bf;
typedef __attribute__((ext_vector_type(8)))  float  v8f;

// ---------------------------------------------------------------------------
// Generic bf16 WMMA GEMM:  C[M,N] = A[M,K] * B[N,K]^T (+bias) (tanh?) (+bf16 copy)
// A row-major lda=K, B row-major ldb=K (B holds the N x K weight matrix).
// Block = 256 threads = 8 waves; each wave owns a 16(M) x 64(N) strip:
//   m0 = blockIdx.y*32 + (wave>>2)*16,  n0 = blockIdx.x*256 + (wave&3)*64.
// Per K-step each wave issues 4 WMMAs sharing ONE A fragment (A amortized 4x,
// ~26 FLOP/byte of fragment traffic; W_out stays L2-resident on the 192MB L2).
// Fragments are loaded directly from global in these lane layouts (A/B share one contiguous k order):
//   A (16-bit 16x32): lane&15 -> row, (lane>>4)*16 -> 16 contiguous K values
//   B (16-bit 32x16): lane&15 -> col, (lane>>4)*16 -> 16 contiguous K values
//   C (32-bit 16x16): vgpr j -> M = j + (lane>>4)*8, N = n0 + (lane&15)
// ---------------------------------------------------------------------------
__global__ __launch_bounds__(256)
void gemm_bf16_wmma(const __bf16* __restrict__ Amat,
                    const __bf16* __restrict__ Bmat,
                    float* __restrict__ C,
                    __bf16* __restrict__ C2,          // optional bf16 mirror (ld = N)
                    const float* __restrict__ bias,   // optional, indexed by col
                    int M, int N, int K, long ldc, int act)
{
    const int tid  = threadIdx.x;
    const int w    = tid >> 5;          // wave id (wave32)
    const int l    = tid & 31;          // lane id
    const int half = l >> 4;
    const int lr   = l & 15;
    const int m0   = blockIdx.y * 32 + (w >> 2) * 16;
    const int n0   = blockIdx.x * 256 + (w & 3) * 64;
    if (m0 >= M || n0 >= N) return;     // wave-uniform guard (EXEC stays all-ones)

    const __bf16* pa = Amat + (size_t)(m0 + lr) * K + half * 16;
    const __bf16* pb = Bmat + (size_t)(n0 + lr) * K + half * 16;
    const size_t  nstep = (size_t)16 * K;   // 16 B-rows (cols of C) per sub-tile

    v8f acc0 = {}, acc1 = {}, acc2 = {}, acc3 = {};
#pragma unroll 2
    for (int k = 0; k < K; k += 32) {
        // prefetch the streaming B operand ahead (sub-tiles 0 and 2 only)
        __builtin_prefetch(pb + k + 256, 0, 0);
        __builtin_prefetch(pb + nstep * 2 + k + 256, 0, 0);

        v8bf alo = *(const v8bf*)(pa + k);        // K = k+off    .. +7
        v8bf ahi = *(const v8bf*)(pa + k + 8);    // K = k+8+off  .. +7
        v16bf a  = __builtin_shufflevector(alo, ahi,
                       0, 1, 2, 3, 4, 5, 6, 7, 8, 9, 10, 11, 12, 13, 14, 15);

        v16bf b0 = *(const v16bf*)(pb + k);
        v16bf b1 = *(const v16bf*)(pb + nstep + k);
        v16bf b2 = *(const v16bf*)(pb + nstep * 2 + k);
        v16bf b3 = *(const v16bf*)(pb + nstep * 3 + k);

        acc0 = __builtin_amdgcn_wmma_f32_16x16x32_bf16(false, a, false, b0,
                                                       (short)0, acc0, false, false);
        acc1 = __builtin_amdgcn_wmma_f32_16x16x32_bf16(false, a, false, b1,
                                                       (short)0, acc1, false, false);
        acc2 = __builtin_amdgcn_wmma_f32_16x16x32_bf16(false, a, false, b2,
                                                       (short)0, acc2, false, false);
        acc3 = __builtin_amdgcn_wmma_f32_16x16x32_bf16(false, a, false, b3,
                                                       (short)0, acc3, false, false);
    }

    const int rbase = m0 + half * 8;
    v8f accs[4] = {acc0, acc1, acc2, acc3};
#pragma unroll
    for (int i = 0; i < 4; ++i) {
        const int col  = n0 + i * 16 + lr;
        const float bv = bias ? bias[col] : 0.0f;
#pragma unroll
        for (int j = 0; j < 8; ++j) {
            float v = accs[i][j] + bv;
            if (act) v = tanhf(v);
            C[(size_t)(rbase + j) * ldc + col] = v;
            if (C2) C2[(size_t)(rbase + j) * N + col] = (__bf16)v;
        }
    }
}

// ---------------------------------------------------------------------------
// Per-step Bahdanau attention + RNN input assembly. One block per batch row.
//   q = h_prev @ W_d2a^T ; e_s = v . tanh(U + q) ; softmax ; ctx = a @ keys
//   cat[b] = [ emb[tok] | ctx | h_prev ]  (bf16)
// ---------------------------------------------------------------------------
__global__ __launch_bounds__(256)
void attn_step(const float* __restrict__ keys,   // [B,S,E] f32
               const float* __restrict__ U,      // [B,S,A] f32
               const float* __restrict__ Wd2a,   // [A,H]   f32
               const float* __restrict__ vw,     // [A]     f32
               const float* __restrict__ emb,    // [V,D]   f32
               const int*   __restrict__ tgt,    // [B,T]
               const float* __restrict__ h,      // [B,H]   f32 (h_prev)
               __bf16*      __restrict__ cat,    // [B,kCat] bf16
               int t)
{
    __shared__ float h_s[kH];
    __shared__ float q_s[kA];
    __shared__ float e_s[kS];
    __shared__ float red[2];
    const int b   = blockIdx.x;
    const int tid = threadIdx.x;

    for (int i = tid; i < kH; i += 256) h_s[i] = h[b * kH + i];
    __syncthreads();

    // q[a]: one attention unit per thread (kA == 256 == blockDim)
    {
        float aq = 0.f;
        const float* wrow = Wd2a + (size_t)tid * kH;
        for (int k = 0; k < kH; ++k) aq += h_s[k] * wrow[k];
        q_s[tid] = aq;
    }
    __syncthreads();

    if (tid < kS) {
        const float* Ub = U + ((size_t)b * kS + tid) * kA;
        float e = 0.f;
        for (int a = 0; a < kA; ++a) e += vw[a] * tanhf(Ub[a] + q_s[a]);
        e_s[tid] = e;
    }
    __syncthreads();
    if (tid == 0) {
        float m = e_s[0];
        for (int s = 1; s < kS; ++s) m = fmaxf(m, e_s[s]);
        red[0] = m;
    }
    __syncthreads();
    if (tid < kS) e_s[tid] = expf(e_s[tid] - red[0]);
    __syncthreads();
    if (tid == 0) {
        float s = 0.f;
        for (int i = 0; i < kS; ++i) s += e_s[i];
        red[1] = 1.0f / s;
    }
    __syncthreads();
    const float inv = red[1];

    __bf16* cb = cat + (size_t)b * kCat;
    // context (f32 accumulation against the original f32 keys)
    for (int j = tid; j < kE; j += 256) {
        float c = 0.f;
        const float* kp = keys + (size_t)b * kS * kE + j;
        for (int s = 0; s < kS; ++s) c += e_s[s] * kp[(size_t)s * kE];
        cb[kD + j] = (__bf16)(c * inv);
    }
    // teacher-forced embedding slice
    const int tok = tgt[b * kT + t];
    if (tid < kD) cb[tid] = (__bf16)emb[(size_t)tok * kD + tid];
    // h_prev slice (for the fused W_hh part of the RNN GEMM)
    for (int j = tid; j < kH; j += 256) cb[kD + kE + j] = (__bf16)h_s[j];
}

// ---------------------------------------------------------------------------
// Small setup kernels
// ---------------------------------------------------------------------------
__global__ void cvt_f32_bf16(const float* __restrict__ s, __bf16* __restrict__ d, long n)
{
    long i = (long)blockIdx.x * blockDim.x + threadIdx.x;
    if (i < n) d[i] = (__bf16)s[i];
}

__global__ void build_wcat(const float* __restrict__ Wih, const float* __restrict__ Whh,
                           __bf16* __restrict__ Wc)
{
    int i = blockIdx.x * blockDim.x + threadIdx.x;     // n = kH * kCat
    if (i >= kH * kCat) return;
    int row = i / kCat, c = i % kCat;
    float v = (c < (kD + kE)) ? Wih[(size_t)row * (kD + kE) + c]
                              : Whh[(size_t)row * kH + (c - kD - kE)];
    Wc[i] = (__bf16)v;
}

__global__ void bias_sum(const float* a, const float* b, float* o, int n)
{
    int i = blockIdx.x * blockDim.x + threadIdx.x;
    if (i < n) o[i] = a[i] + b[i];
}

__global__ void zero_f32(float* p, int n)
{
    int i = blockIdx.x * blockDim.x + threadIdx.x;
    if (i < n) p[i] = 0.f;
}

// ---------------------------------------------------------------------------
extern "C" void kernel_launch(void* const* d_in, const int* in_sizes, int n_in,
                              void* d_out, int out_size, void* d_ws, size_t ws_size,
                              hipStream_t stream)
{
    const float* enc  = (const float*)d_in[0];   // [B,S,E]
    const int*   tgt  = (const int*)  d_in[1];   // [B,T]
    const float* We2a = (const float*)d_in[2];   // [A,E]
    const float* Wd2a = (const float*)d_in[3];   // [A,H]
    const float* vw   = (const float*)d_in[4];   // [A]
    const float* emb  = (const float*)d_in[5];   // [V,D]
    const float* Wih  = (const float*)d_in[6];   // [H,D+E]
    const float* bih  = (const float*)d_in[7];   // [H]
    const float* Whh  = (const float*)d_in[8];   // [H,H]
    const float* bhh  = (const float*)d_in[9];   // [H]
    const float* Wout = (const float*)d_in[10];  // [V,H]
    const float* bout = (const float*)d_in[11];  // [V]
    float* out = (float*)d_out;                  // [B,T,V]

    // ---- workspace carve (256B aligned) ----
    char* ws = (char*)d_ws;
    size_t off = 0;
    auto carve = [&](size_t bytes) -> char* {
        char* p = ws + off;
        off = (off + bytes + 255) & ~(size_t)255;
        return p;
    };
    __bf16* keys_bf = (__bf16*)carve((size_t)kB * kS * kE * 2);   // 4 MB
    __bf16* We2a_bf = (__bf16*)carve((size_t)kA * kE * 2);        // 256 KB
    __bf16* Wout_bf = (__bf16*)carve((size_t)kV * kH * 2);        // 32.8 MB (L2-resident)
    __bf16* Wcat_bf = (__bf16*)carve((size_t)kH * kCat * 2);      // 1.3 MB
    float*  U       = (float*)carve((size_t)kB * kS * kA * 4);    // 4 MB
    float*  bsum    = (float*)carve((size_t)kH * 4);
    float*  h       = (float*)carve((size_t)kB * kH * 4);
    __bf16* h_bf    = (__bf16*)carve((size_t)kB * kH * 2);
    __bf16* cat     = (__bf16*)carve((size_t)kB * kCat * 2);

    // ---- one-time setup ----
    {
        long n;
        n = (long)kB * kS * kE;
        cvt_f32_bf16<<<(unsigned)((n + 255) / 256), 256, 0, stream>>>(enc, keys_bf, n);
        n = (long)kA * kE;
        cvt_f32_bf16<<<(unsigned)((n + 255) / 256), 256, 0, stream>>>(We2a, We2a_bf, n);
        n = (long)kV * kH;
        cvt_f32_bf16<<<(unsigned)((n + 255) / 256), 256, 0, stream>>>(Wout, Wout_bf, n);
        int nc = kH * kCat;
        build_wcat<<<(nc + 255) / 256, 256, 0, stream>>>(Wih, Whh, Wcat_bf);
        bias_sum<<<(kH + 255) / 256, 256, 0, stream>>>(bih, bhh, bsum, kH);
        zero_f32<<<(kB * kH + 255) / 256, 256, 0, stream>>>(h, kB * kH);
    }

    // U = keys @ W_e2a^T  : [4096,512] x [256,512]^T -> [4096,256]
    gemm_bf16_wmma<<<dim3(kA / 256, (kB * kS) / 32), 256, 0, stream>>>(
        keys_bf, We2a_bf, U, nullptr, nullptr,
        kB * kS, kA, kE, (long)kA, 0);

    // ---- recurrent decode, T = 48 steps ----
    for (int t = 0; t < kT; ++t) {
        // attention + build [x_t | ctx | h_prev]
        attn_step<<<kB, 256, 0, stream>>>(enc, U, Wd2a, vw, emb, tgt, h, cat, t);

        // h_new = tanh(cat @ Wcat^T + bsum)  -> h (f32) and h_bf (bf16)
        gemm_bf16_wmma<<<dim3(kH / 256, 1), 256, 0, stream>>>(
            cat, Wcat_bf, h, h_bf, bsum,
            kB, kH, kCat, (long)kH, 1);

        // logits[:, t, :] = h_new @ W_out^T + b_out   (ldc = T*V into [B,T,V])
        gemm_bf16_wmma<<<dim3(kV / 256, 1), 256, 0, stream>>>(
            h_bf, Wout_bf, out + (size_t)t * kV, nullptr, bout,
            kB, kV, kH, (long)kT * kV, 0);
    }
}
